// KnowledgeGraphEmbedder_9328668966985
// MI455X (gfx1250) — matrix operand order repaired and checked by reading, not compiled, not run on hardware
//
#include <hip/hip_runtime.h>
#include <hip/hip_bf16.h>

#define NN 200000   // nodes
#define NE 200000   // edges per relation
#define NR 7        // relations
#define LN_EPS 1e-5f

typedef __attribute__((ext_vector_type(16))) __bf16       v16bf;
typedef __attribute__((ext_vector_type(8)))  float        v8f;
typedef __attribute__((ext_vector_type(4)))  unsigned int u32x4;

union AF { v16bf v; u32x4 u[2]; };

__device__ __forceinline__ unsigned short f2bf(float f) {
  unsigned int u = __float_as_uint(f);
  u += 0x7fffu + ((u >> 16) & 1u);     // round-to-nearest-even
  return (unsigned short)(u >> 16);
}

// ---------------------------------------------------------------------------
// W_r = sum_b coeff[r,b] * bases[b]  -> store bf16 TRANSPOSED: Wt[r][o][i]
// ---------------------------------------------------------------------------
__global__ void relw_kernel(const float* __restrict__ bases,
                            const float* __restrict__ coeff,
                            unsigned short* __restrict__ Wt,
                            int d_in, int d_out) {
  long long idx = (long long)blockIdx.x * blockDim.x + threadIdx.x;
  long long per = (long long)d_in * d_out;
  if (idx >= (long long)NR * per) return;
  int r  = (int)(idx / per);
  int io = (int)(idx % per);
  int i  = io / d_out;
  int o  = io % d_out;
  float w = 0.f;
#pragma unroll
  for (int b = 0; b < 4; ++b)
    w += coeff[r * 4 + b] * bases[(long long)b * per + (long long)i * d_out + o];
  Wt[(long long)r * per + (long long)o * d_in + i] = f2bf(w);
}

// w0 [d_in,d_out] f32 -> W0t [d_out,d_in] bf16
__global__ void w0t_kernel(const float* __restrict__ w0,
                           unsigned short* __restrict__ W0t,
                           int d_in, int d_out) {
  int idx = blockIdx.x * blockDim.x + threadIdx.x;
  if (idx >= d_in * d_out) return;
  int i = idx / d_out, o = idx % d_out;
  W0t[o * d_in + i] = f2bf(w0[(long long)i * d_out + o]);
}

// f32 -> bf16 elementwise
__global__ void x2bf_kernel(const float* __restrict__ x,
                            unsigned short* __restrict__ xb, long long n) {
  long long idx = (long long)blockIdx.x * blockDim.x + threadIdx.x;
  if (idx < n) xb[idx] = f2bf(x[idx]);
}

// ---------------------------------------------------------------------------
// Fused scatter + degree:  acc[dst[e]][:] += x[src[e]][:] ; deg[dst[e]] += 1
// One thread per (edge, 4-channel group).
// ---------------------------------------------------------------------------
__global__ void scatter_kernel(const float* __restrict__ xf,
                               const int* __restrict__ src,
                               const int* __restrict__ dst,
                               float* __restrict__ acc,
                               float* __restrict__ deg,
                               int E, int dq /* d/4 */) {
  long long idx = (long long)blockIdx.x * blockDim.x + threadIdx.x;
  if (idx >= (long long)E * dq) return;
  int e = (int)(idx / dq);
  int q = (int)(idx % dq);
  int s = src[e], t = dst[e];
  float4 v = ((const float4*)(xf + (long long)s * dq * 4))[q];
  float* ar = acc + (long long)t * dq * 4 + q * 4;
  atomicAdd(ar + 0, v.x);
  atomicAdd(ar + 1, v.y);
  atomicAdd(ar + 2, v.z);
  atomicAdd(ar + 3, v.w);
  if (q == 0) atomicAdd(&deg[t], 1.0f);
}

// accb = bf16( acc / max(deg,1) ), 4 elems/thread
__global__ void norm_cvt_kernel(const float* __restrict__ acc,
                                const float* __restrict__ deg,
                                unsigned short* __restrict__ accb,
                                int N, int dq /* d/4 */) {
  long long idx = (long long)blockIdx.x * blockDim.x + threadIdx.x;
  if (idx >= (long long)N * dq) return;
  int n = (int)(idx / dq);
  float dg = deg[n];
  float r  = 1.0f / (dg > 1.0f ? dg : 1.0f);
  float4 v = ((const float4*)acc)[idx];
  uint2 o;
  o.x = (unsigned)f2bf(v.x * r) | ((unsigned)f2bf(v.y * r) << 16);
  o.y = (unsigned)f2bf(v.z * r) | ((unsigned)f2bf(v.w * r) << 16);
  ((uint2*)accb)[idx] = o;
}

// ---------------------------------------------------------------------------
// GEMM: C[M,Nout] (f32) (+)= A[M,K](bf16,row-major) x Bt[Nout,K]^T (bf16)
// Block = 256 thr = 8 waves; block tile = 512 rows x 64 cols.
// Weight panel (64 x K, contiguous) staged in LDS once; each wave does
// 4 M-tiles so every ds-loaded B fragment feeds 4 WMMAs. A fragments are
// double-buffered across the K loop to hide global latency behind WMMAs.
// ---------------------------------------------------------------------------
__global__ __launch_bounds__(256)
void gemm_bf16_wmma(const unsigned short* __restrict__ A,
                    const unsigned short* __restrict__ Bt,
                    const float* __restrict__ bias,
                    float* __restrict__ C,
                    int M, int K, int Nout, int accumulate) {
  __shared__ unsigned short lB[64 * 256];      // up to 32 KB (K<=256)
  const int tid  = threadIdx.x;
  const int wave = tid >> 5;
  const int lane = tid & 31;
  const int l16  = lane & 15;
  const int hig  = lane >> 4;                  // 0/1: K-half select
  const int n0   = blockIdx.y * 64;

  // stage 64 x K weight panel (contiguous at Bt + n0*K) into LDS
  {
    const u32x4* sp = (const u32x4*)(Bt + (long long)n0 * K);
    u32x4*       dp = (u32x4*)lB;
    const int nchunk = (64 * K) >> 3;          // u32x4 = 8 halves
    for (int i = tid; i < nchunk; i += 256) dp[i] = sp[i];
  }
  __syncthreads();

  // 4 M-tiles per wave (clamped; duplicate tiles write identical values)
  const unsigned short* arow[4];
  int m0[4];
#pragma unroll
  for (int t = 0; t < 4; ++t) {
    int m = (blockIdx.x * 32 + wave * 4 + t) * 16;
    if (m > M - 16) m = M - 16;
    m0[t] = m;
    arow[t] = A + (long long)(m + l16) * K;
  }

  v8f c[4][4];                                 // [tile][j]
  if (accumulate) {
#pragma unroll
    for (int t = 0; t < 4; ++t)
#pragma unroll
      for (int v = 0; v < 8; ++v) {
        const float* crow = C + (long long)(m0[t] + hig * 8 + v) * Nout + n0 + l16;
        c[t][0][v] = crow[0];  c[t][1][v] = crow[16];
        c[t][2][v] = crow[32]; c[t][3][v] = crow[48];
      }
  } else {
    float b0 = bias[n0 + l16],      b1 = bias[n0 + 16 + l16];
    float b2 = bias[n0 + 32 + l16], b3 = bias[n0 + 48 + l16];
#pragma unroll
    for (int t = 0; t < 4; ++t)
#pragma unroll
      for (int v = 0; v < 8; ++v) {
        c[t][0][v] = b0; c[t][1][v] = b1; c[t][2][v] = b2; c[t][3][v] = b3;
      }
  }

  // prefetch first A fragments
  AF aN[4];
#pragma unroll
  for (int t = 0; t < 4; ++t) {
    aN[t].u[0] = *(const u32x4*)(arow[t] + hig * 8);
    aN[t].u[1] = *(const u32x4*)(arow[t] + 16 + hig * 8);
  }

  for (int kk = 0; kk < K; kk += 32) {
    AF aC[4];
#pragma unroll
    for (int t = 0; t < 4; ++t) aC[t].v = aN[t].v;
    if (kk + 32 < K) {
#pragma unroll
      for (int t = 0; t < 4; ++t) {
        aN[t].u[0] = *(const u32x4*)(arow[t] + kk + 32 + hig * 8);
        aN[t].u[1] = *(const u32x4*)(arow[t] + kk + 48 + hig * 8);
      }
    }
#pragma unroll
    for (int j = 0; j < 4; ++j) {
      AF b;
      const u32x4* bp = (const u32x4*)(lB + (j * 16 + l16) * K + kk + hig * 8);
      b.u[0] = bp[0];
      b.u[1] = bp[2];
#pragma unroll
      for (int t = 0; t < 4; ++t)
        c[t][j] = __builtin_amdgcn_wmma_f32_16x16x32_bf16(
            false, aC[t].v, false, b.v, (short)0, c[t][j], false, false);
    }
  }

#pragma unroll
  for (int t = 0; t < 4; ++t)
#pragma unroll
    for (int v = 0; v < 8; ++v) {
      float* crow = C + (long long)(m0[t] + hig * 8 + v) * Nout + n0 + l16;
      crow[0]  = c[t][0][v]; crow[16] = c[t][1][v];
      crow[32] = c[t][2][v]; crow[48] = c[t][3][v];
    }
}

// ---------------------------------------------------------------------------
// a = silu(h); LayerNorm over last dim; write f32 (+ optional bf16 copy).
// One wave (32 lanes) per row; d in {128,256}.
// ---------------------------------------------------------------------------
__global__ __launch_bounds__(256)
void silu_ln_kernel(const float* __restrict__ h,
                    const float* __restrict__ g,
                    const float* __restrict__ be,
                    float* __restrict__ outf,
                    unsigned short* __restrict__ outb,
                    int N, int d) {
  const int wave = threadIdx.x >> 5;
  const int lane = threadIdx.x & 31;
  const int row  = blockIdx.x * 8 + wave;
  if (row >= N) return;
  const float* hr = h + (long long)row * d;
  const int per = d >> 5;                  // elems per lane (4 or 8)
  float vals[8];
  float s = 0.f, s2 = 0.f;
  for (int i = 0; i < per; ++i) {
    float x = hr[lane + i * 32];
    float a = x / (1.0f + __expf(-x));     // silu
    vals[i] = a; s += a; s2 += a * a;
  }
  for (int m = 16; m >= 1; m >>= 1) {
    s  += __shfl_xor(s,  m, 32);
    s2 += __shfl_xor(s2, m, 32);
  }
  float inv = 1.0f / (float)d;
  float mu  = s * inv;
  float var = s2 * inv - mu * mu;
  float rs  = rsqrtf(var + LN_EPS);
  for (int i = 0; i < per; ++i) {
    int c = lane + i * 32;
    float o = (vals[i] - mu) * rs * g[c] + be[c];
    outf[(long long)row * d + c] = o;
    if (outb) outb[(long long)row * d + c] = f2bf(o);
  }
}

// ---------------------------------------------------------------------------
// Host orchestration
// ---------------------------------------------------------------------------
static void run_layer(const float* xf, const unsigned short* xb,
                      const float* bases, const float* coeff,
                      const float* w0, const float* b0,
                      const float* g, const float* be,
                      const int* src, const int* dst,
                      int d_in, int d_out,
                      float* h, float* acc, float* deg,
                      unsigned short* accb,
                      unsigned short* Wt, unsigned short* W0t,
                      float* outf, unsigned short* outb,
                      hipStream_t stream) {
  const int B = 256;
  long long relw_n = (long long)NR * d_in * d_out;
  relw_kernel<<<(unsigned)((relw_n + B - 1) / B), B, 0, stream>>>(bases, coeff, Wt, d_in, d_out);
  w0t_kernel<<<(d_in * d_out + B - 1) / B, B, 0, stream>>>(w0, W0t, d_in, d_out);

  // self loop: h = xb @ w0 + b0
  const int mtiles = NN / 16;                        // 12500
  dim3 gg((mtiles + 31) / 32, d_out / 64);           // 512-row x 64-col blocks
  gemm_bf16_wmma<<<gg, B, 0, stream>>>(xb, W0t, b0, h, NN, d_in, d_out, 0);

  const int dq = d_in / 4;
  long long ndq = (long long)NN * dq;
  long long edq = (long long)NE * dq;
  for (int r = 0; r < NR; ++r) {
    hipMemsetAsync(deg, 0, (size_t)NN * sizeof(float), stream);
    hipMemsetAsync(acc, 0, (size_t)NN * d_in * sizeof(float), stream);
    scatter_kernel<<<(unsigned)((edq + B - 1) / B), B, 0, stream>>>(
        xf, src + (long long)r * NE, dst + (long long)r * NE, acc, deg, NE, dq);
    norm_cvt_kernel<<<(unsigned)((ndq + B - 1) / B), B, 0, stream>>>(acc, deg, accb, NN, dq);
    // h += (acc/deg) @ W_r
    gemm_bf16_wmma<<<gg, B, 0, stream>>>(accb, Wt + (long long)r * d_in * d_out,
                                         nullptr, h, NN, d_in, d_out, 1);
  }
  silu_ln_kernel<<<(NN + 7) / 8, B, 0, stream>>>(h, g, be, outf, outb, NN, d_out);
}

extern "C" void kernel_launch(void* const* d_in, const int* in_sizes, int n_in,
                              void* d_out, int out_size, void* d_ws, size_t ws_size,
                              hipStream_t stream) {
  const float* x   = (const float*)d_in[0];
  const int*   src = (const int*)d_in[1];
  const int*   dst = (const int*)d_in[2];
  const float* prm[3][6];
  for (int L = 0; L < 3; ++L)
    for (int k = 0; k < 6; ++k)
      prm[L][k] = (const float*)d_in[3 + L * 6 + k];
  // prm[L] = { bases, coeff, w0, b0, g, be }

  // workspace carve (all 256B aligned)
  size_t off = 0;
  auto carve = [&](size_t bytes) -> void* {
    void* p = (char*)d_ws + off;
    off += (bytes + 255) & ~(size_t)255;
    return p;
  };
  float*          h    = (float*)carve((size_t)NN * 256 * 4);
  float*          acc  = (float*)carve((size_t)NN * 256 * 4);
  float*          xf   = (float*)carve((size_t)NN * 256 * 4);
  unsigned short* xb   = (unsigned short*)carve((size_t)NN * 256 * 2);
  unsigned short* accb = (unsigned short*)carve((size_t)NN * 256 * 2);
  float*          deg  = (float*)carve((size_t)NN * 4);
  unsigned short* Wt   = (unsigned short*)carve((size_t)NR * 256 * 256 * 2);
  unsigned short* W0t  = (unsigned short*)carve((size_t)256 * 256 * 2);

  const int B = 256;
  // layer 0 input: x (f32, d=128) -> bf16 copy
  long long n0 = (long long)NN * 128;
  x2bf_kernel<<<(unsigned)((n0 + B - 1) / B), B, 0, stream>>>(x, xb, n0);

  // layer 0: 128 -> 256 ; LN writes next input (xf f32 + xb bf16)
  run_layer(x, xb, prm[0][0], prm[0][1], prm[0][2], prm[0][3], prm[0][4], prm[0][5],
            src, dst, 128, 256, h, acc, deg, accb, Wt, W0t, xf, xb, stream);
  // layer 1: 256 -> 256
  run_layer(xf, xb, prm[1][0], prm[1][1], prm[1][2], prm[1][3], prm[1][4], prm[1][5],
            src, dst, 256, 256, h, acc, deg, accb, Wt, W0t, xf, xb, stream);
  // layer 2: 256 -> 128 ; final LN writes d_out (f32), no bf16 copy needed
  run_layer(xf, xb, prm[2][0], prm[2][1], prm[2][2], prm[2][3], prm[2][4], prm[2][5],
            src, dst, 256, 128, h, acc, deg, accb, Wt, W0t,
            (float*)d_out, nullptr, stream);
}
